// TransformerBlock_54090818126673
// MI455X (gfx1250) — matrix operand order repaired and checked by reading, not compiled, not run on hardware
//
#include <hip/hip_runtime.h>

// ---------------------------------------------------------------------------
// Shapes (fixed by the reference): B=2 S=2048 D=256 H=8 E=256 F=1024
// ---------------------------------------------------------------------------
constexpr int Bc = 2, Sc = 2048, Dc = 256, Hc = 8, Ec = 256, Fc = 1024;
constexpr int HE = Hc * Ec;          // 2048
constexpr int Rows = Bc * Sc;        // 4096 token rows
constexpr float EPS = 1e-6f;

typedef unsigned short u16;
typedef __attribute__((ext_vector_type(16))) __bf16 v16bf;
typedef __attribute__((ext_vector_type(8)))  float  v8f;

union ABFrag  { uint4 q[2]; v16bf v; };
union CvtFrag { u16  s[16]; v16bf v; };

__device__ inline u16 f2b(float f) {
    unsigned u = __float_as_uint(f);
    return (u16)((u + 0x7FFFu + ((u >> 16) & 1u)) >> 16);   // round-nearest-even
}

__device__ inline v8f wmma_bf16(v16bf a, v16bf b, v8f c) {
    // v_wmma_f32_16x16x32_bf16  (neg_a, A, neg_b, B, c_mod, C, reuse_a, reuse_b)
    return __builtin_amdgcn_wmma_f32_16x16x32_bf16(false, a, false, b, (short)0, c,
                                                   false, false);
}

// A fragment: 16x32 bf16, row-major source, lda in elements.
// lanes 0-15: M=lane, K=0..7 (v0..3) and 16..23 (v4..7); lanes 16-31: K=+8.
__device__ inline v16bf load_a_frag(const u16* A, int lda, int k0, int lane) {
    int row = lane & 15;
    int kb  = (lane >> 4) * 8;
    const u16* p = A + (size_t)row * lda + k0 + kb;
    ABFrag f;
    f.q[0] = *reinterpret_cast<const uint4*>(p);        // K kb..kb+7
    f.q[1] = *reinterpret_cast<const uint4*>(p + 16);   // K kb+16..kb+23
    return f.v;
}

// B fragment from B^T[N,K] row-major: lanes 0-15: N=lane, K=0..7 and 16..23;
// lanes 16-31: K=+8.
__device__ inline v16bf load_b_frag(const u16* Bt, int ldb, int n0, int k0, int lane) {
    int col = lane & 15;
    int kb  = (lane >> 4) * 8;
    const u16* p = Bt + (size_t)(n0 + col) * ldb + k0 + kb;
    ABFrag f;
    f.q[0] = *reinterpret_cast<const uint4*>(p);        // K kb..kb+7
    f.q[1] = *reinterpret_cast<const uint4*>(p + 16);   // K kb+16..kb+23
    return f.v;
}

__device__ inline float gelu_tanh(float x) {
    float x3 = x * x * x;
    return 0.5f * x * (1.0f + tanhf(0.7978845608028654f * (x + 0.044715f * x3)));
}

// ---------------------------------------------------------------------------
// Transpose + fp32 -> bf16 convert:  in[R,C] -> out[C,R]
// ---------------------------------------------------------------------------
__global__ void tc_kernel(const float* __restrict__ in, u16* __restrict__ out,
                          int R, int C) {
    int i = blockIdx.x * 256 + threadIdx.x;
    if (i < R * C) {
        int r = i / C, c = i - r * C;
        out[(size_t)c * R + r] = f2b(in[i]);
    }
}

// ---------------------------------------------------------------------------
// LayerNorm over last dim (D=256), one row per 256-thread block, bf16 out
// ---------------------------------------------------------------------------
__global__ __launch_bounds__(256)
void ln_kernel(const float* __restrict__ x, const float* __restrict__ scale,
               const float* __restrict__ bias, u16* __restrict__ out) {
    __shared__ float red[256];
    int row = blockIdx.x, t = threadIdx.x;
    float v = x[(size_t)row * Dc + t];
    red[t] = v; __syncthreads();
    for (int s = 128; s > 0; s >>= 1) {
        if (t < s) red[t] += red[t + s];
        __syncthreads();
    }
    float mean = red[0] * (1.0f / Dc); __syncthreads();
    float xc = v - mean;
    red[t] = xc * xc; __syncthreads();
    for (int s = 128; s > 0; s >>= 1) {
        if (t < s) red[t] += red[t + s];
        __syncthreads();
    }
    float inv = rsqrtf(red[0] * (1.0f / Dc) + EPS);
    out[(size_t)row * Dc + t] = f2b(xc * inv * scale[t] + bias[t]);
}

// ---------------------------------------------------------------------------
// Generic bf16 WMMA GEMM:  C[M,N] = A[M,K] * B^T[N,K]^T  (+ epilogue)
// 128 threads = 4 waves; each wave owns one 16x16 C tile (block covers 16x64).
// EPI: 0 = bf16 +bias; 1 = gelu->bf16 +bias; 2 = fp32 +bias +residual;
//      3 = bf16 +bias, scatter-store as V^T[B,H,E,S]
// ---------------------------------------------------------------------------
template <int EPI>
__global__ __launch_bounds__(128)
void gemm_bf16(const u16* __restrict__ A, const u16* __restrict__ Bt,
               void* __restrict__ out, const float* __restrict__ bias,
               const float* __restrict__ res,
               int K, int lda, int ldb, int ldc) {
    int lane = threadIdx.x & 31;
    int wave = threadIdx.x >> 5;
    int m0 = blockIdx.y * 16;
    int n0 = (blockIdx.x * 4 + wave) * 16;

    const u16* Arow = A + (size_t)m0 * lda;
    v8f c = {};
    for (int k0 = 0; k0 < K; k0 += 32) {
        v16bf a = load_a_frag(Arow, lda, k0, lane);
        v16bf b = load_b_frag(Bt, ldb, n0, k0, lane);
        c = wmma_bf16(a, b, c);
    }

    int hi  = lane >> 4;
    int n   = n0 + (lane & 15);
    float bv = bias ? bias[n] : 0.0f;
#pragma unroll
    for (int r = 0; r < 8; ++r) {
        int m = m0 + r + hi * 8;
        float acc = c[r] + bv;
        if (EPI == 0) {
            ((u16*)out)[(size_t)m * ldc + n] = f2b(acc);
        } else if (EPI == 1) {
            ((u16*)out)[(size_t)m * ldc + n] = f2b(gelu_tanh(acc));
        } else if (EPI == 2) {
            size_t idx = (size_t)m * ldc + n;
            ((float*)out)[idx] = acc + res[idx];
        } else { // EPI == 3 : V^T[B,H,E,S]
            int b_ = m >> 11, s = m & (Sc - 1);
            int hh = n >> 8,  e = n & (Ec - 1);
            ((u16*)out)[((((size_t)b_ * Hc + hh) * Ec + e) << 11) + s] = f2b(acc);
        }
    }
}

// ---------------------------------------------------------------------------
// Attention: one block (4 waves) per 16-query tile per (b,h).
// Scores 16x2048 fp32 in LDS (128 KB, fits CDNA5 320 KB WGP LDS),
// softmax in fp32, attn written to output, then P*V via WMMA.
// ---------------------------------------------------------------------------
__global__ __launch_bounds__(128)
void attn_kernel(const u16* __restrict__ qb, const u16* __restrict__ kb,
                 const u16* __restrict__ vT, float* __restrict__ attn_out,
                 u16* __restrict__ valb) {
    __shared__ float sc[16 * Sc];     // 128 KB scores
    __shared__ float red[128];
    __shared__ float rowmax[16], rowsum[16];

    int lane = threadIdx.x & 31;
    int wave = threadIdx.x >> 5;
    int b = blockIdx.z, h = blockIdx.y, q0 = blockIdx.x * 16;

    const u16* Aq = qb + ((size_t)(b * Sc + q0)) * HE + h * Ec;   // [16,256] lda=HE
    const u16* Bk = kb + ((size_t)b * Sc) * HE + h * Ec;          // keys as B^T, ldb=HE

    // ---- scores = q * k^T / sqrt(E) -------------------------------------
    for (int nt = wave; nt < Sc / 16; nt += 4) {
        v8f c = {};
        for (int k0 = 0; k0 < Ec; k0 += 32) {
            v16bf a = load_a_frag(Aq, HE, k0, lane);
            v16bf bf = load_b_frag(Bk, HE, nt * 16, k0, lane);
            c = wmma_bf16(a, bf, c);
        }
        int hi = lane >> 4, col = nt * 16 + (lane & 15);
#pragma unroll
        for (int r = 0; r < 8; ++r)
            sc[(r + hi * 8) * Sc + col] = c[r] * 0.0625f;   // 1/sqrt(256)
    }
    __syncthreads();

    // ---- softmax over 2048 keys: 8 threads per row ----------------------
    int row = threadIdx.x >> 3, seg = threadIdx.x & 7;
    float m = -3.0e38f;
    for (int i = 0; i < Sc / 8; ++i) {
        float x = sc[row * Sc + seg + i * 8];
        m = fmaxf(m, x);
    }
    red[threadIdx.x] = m; __syncthreads();
    if (seg == 0) {
        float mm = red[row * 8];
        for (int j = 1; j < 8; ++j) mm = fmaxf(mm, red[row * 8 + j]);
        rowmax[row] = mm;
    }
    __syncthreads();
    m = rowmax[row];
    float sum = 0.0f;
    for (int i = 0; i < Sc / 8; ++i) {
        int idx = row * Sc + seg + i * 8;
        float e = __expf(sc[idx] - m);
        sc[idx] = e;
        sum += e;
    }
    red[threadIdx.x] = sum; __syncthreads();
    if (seg == 0) {
        float ss = 0.0f;
        for (int j = 0; j < 8; ++j) ss += red[row * 8 + j];
        rowsum[row] = ss;
    }
    __syncthreads();
    float inv = 1.0f / rowsum[row];
    float* arow = attn_out + (((size_t)(b * Hc + h) * Sc + (q0 + row)) * Sc);
    for (int i = 0; i < Sc / 8; ++i) {
        int cidx = seg + i * 8;
        float p = sc[row * Sc + cidx] * inv;
        sc[row * Sc + cidx] = p;
        arow[cidx] = p;
    }
    __syncthreads();

    // ---- val = P * V  (A from LDS fp32->bf16, B = V^T[E,S] from global) --
    const u16* Vb = vT + ((size_t)(b * Hc + h) * Ec) * Sc;   // [256, 2048] ldb=Sc
    for (int nt = wave; nt < Ec / 16; nt += 4) {
        v8f c = {};
        int arow_l = lane & 15;
        int kbA = (lane >> 4) * 8;
        for (int k0 = 0; k0 < Sc; k0 += 32) {
            const float* pa = &sc[arow_l * Sc + k0 + kbA];
            CvtFrag f;
#pragma unroll
            for (int i = 0; i < 8; ++i) f.s[i]     = f2b(pa[i]);
#pragma unroll
            for (int i = 0; i < 8; ++i) f.s[8 + i] = f2b(pa[16 + i]);
            v16bf bf = load_b_frag(Vb, Sc, nt * 16, k0, lane);
            c = wmma_bf16(f.v, bf, c);
        }
        int hi = lane >> 4;
        int colg = h * Ec + nt * 16 + (lane & 15);
#pragma unroll
        for (int r = 0; r < 8; ++r) {
            int mrow = q0 + r + hi * 8;
            valb[(size_t)(b * Sc + mrow) * HE + colg] = f2b(c[r]);
        }
    }
}

// ---------------------------------------------------------------------------
// Host-side orchestration
// ---------------------------------------------------------------------------
extern "C" void kernel_launch(void* const* d_in, const int* in_sizes, int n_in,
                              void* d_out, int out_size, void* d_ws, size_t ws_size,
                              hipStream_t stream) {
    const float* x         = (const float*)d_in[0];
    const float* ln1_scale = (const float*)d_in[1];
    const float* ln1_bias  = (const float*)d_in[2];
    const float* wq = (const float*)d_in[3];  const float* bq = (const float*)d_in[4];
    const float* wk = (const float*)d_in[5];  const float* bk = (const float*)d_in[6];
    const float* wv = (const float*)d_in[7];  const float* bv = (const float*)d_in[8];
    const float* wo = (const float*)d_in[9];  const float* bo = (const float*)d_in[10];
    const float* ln2_scale = (const float*)d_in[11];
    const float* ln2_bias  = (const float*)d_in[12];
    const float* w1 = (const float*)d_in[13]; const float* b1 = (const float*)d_in[14];
    const float* w2 = (const float*)d_in[15]; const float* b2 = (const float*)d_in[16];

    float* out_x    = (float*)d_out;                         // [B,S,D]
    float* out_attn = (float*)d_out + (size_t)Rows * Dc;     // [B,H,S,S]

    // workspace carve-up (256B aligned)
    char* p = (char*)d_ws;
    auto take = [&](size_t bytes) { char* r = p; p += (bytes + 255) & ~(size_t)255; return r; };
    u16* wqT  = (u16*)take((size_t)HE * Dc * 2);
    u16* wkT  = (u16*)take((size_t)HE * Dc * 2);
    u16* wvT  = (u16*)take((size_t)HE * Dc * 2);
    u16* woT  = (u16*)take((size_t)Dc * HE * 2);
    u16* w1T  = (u16*)take((size_t)Fc * Dc * 2);
    u16* w2T  = (u16*)take((size_t)Dc * Fc * 2);
    u16* xln  = (u16*)take((size_t)Rows * Dc * 2);
    u16* qb   = (u16*)take((size_t)Rows * HE * 2);
    u16* kbuf = (u16*)take((size_t)Rows * HE * 2);
    u16* vT   = (u16*)take((size_t)Rows * HE * 2);
    u16* valb = (u16*)take((size_t)Rows * HE * 2);
    float* x2 = (float*)take((size_t)Rows * Dc * 4);
    u16* x2ln = (u16*)take((size_t)Rows * Dc * 2);
    u16* hb   = (u16*)take((size_t)Rows * Fc * 2);

    auto tc = [&](const float* in, u16* out, int R, int C) {
        tc_kernel<<<(R * C + 255) / 256, 256, 0, stream>>>(in, out, R, C);
    };
    // weight transposes -> bf16 B^T[N,K]
    tc(wq, wqT, Dc, HE);
    tc(wk, wkT, Dc, HE);
    tc(wv, wvT, Dc, HE);
    tc(wo, woT, HE, Dc);
    tc(w1, w1T, Dc, Fc);
    tc(w2, w2T, Fc, Dc);

    // LN1
    ln_kernel<<<Rows, 256, 0, stream>>>(x, ln1_scale, ln1_bias, xln);

    // QKV projections (M=4096, N=2048, K=256)
    dim3 gQKV(HE / 64, Rows / 16);
    gemm_bf16<0><<<gQKV, 128, 0, stream>>>(xln, wqT, qb,   bq, nullptr, Dc, Dc, Dc, HE);
    gemm_bf16<0><<<gQKV, 128, 0, stream>>>(xln, wkT, kbuf, bk, nullptr, Dc, Dc, Dc, HE);
    gemm_bf16<3><<<gQKV, 128, 0, stream>>>(xln, wvT, vT,   bv, nullptr, Dc, Dc, Dc, 0);

    // attention (scores + softmax + P*V)
    dim3 gA(Sc / 16, Hc, Bc);
    attn_kernel<<<gA, 128, 0, stream>>>(qb, kbuf, vT, out_attn, valb);

    // output projection + residual: x2 = x + val @ wo + bo  (M=4096,N=256,K=2048)
    dim3 gO(Dc / 64, Rows / 16);
    gemm_bf16<2><<<gO, 128, 0, stream>>>(valb, woT, x2, bo, x, HE, HE, HE, Dc);

    // LN2
    ln_kernel<<<Rows, 256, 0, stream>>>(x2, ln2_scale, ln2_bias, x2ln);

    // MLP1: h = gelu(x2ln @ w1 + b1)  (M=4096,N=1024,K=256)
    dim3 gM1(Fc / 64, Rows / 16);
    gemm_bf16<1><<<gM1, 128, 0, stream>>>(x2ln, w1T, hb, b1, nullptr, Dc, Dc, Dc, Fc);

    // MLP2 + residual -> final output  (M=4096,N=256,K=1024)
    dim3 gM2(Dc / 64, Rows / 16);
    gemm_bf16<2><<<gM2, 128, 0, stream>>>(hb, w2T, out_x, b2, x2, Fc, Fc, Fc, Dc);
}
